// MoELayer_19808389169318
// MI455X (gfx1250) — matrix operand order repaired and checked by reading, not compiled, not run on hardware
//
#include <hip/hip_runtime.h>
#include <math.h>

#define H_DIM   1024
#define E_NUM   8
#define FFN_DIM 4096
#define T_TOK   4096
#define KT      64   // K tile staged in LDS per iteration (2 WMMA sub-steps)
#define AS      72   // LDS row stride in bf16 elems: 64 + 8 pad, keeps 16B alignment

typedef __bf16 bf16;
typedef __attribute__((ext_vector_type(16))) __bf16 v16bf;
typedef __attribute__((ext_vector_type(8)))  float  v8f;

struct __align__(16) U4 { unsigned x, y, z, w; };
struct __align__(8)  U2 { unsigned x, y; };
struct __align__(16) F4 { float x, y, z, w; };

union Frag16 { v16bf v; U4 q[2]; };

static __device__ inline unsigned pk_bf16(float a, float b) {
    union { bf16 h[2]; unsigned u; } t;
    t.h[0] = (bf16)a; t.h[1] = (bf16)b;
    return t.u;
}

static __device__ inline void st_bf16x4(bf16* dst, const F4& v) {
    U2 p; p.x = pk_bf16(v.x, v.y); p.y = pk_bf16(v.z, v.w);
    *(U2*)dst = p;
}

// ---------------------------------------------------------------- init
__global__ void moe_init(int* __restrict__ cnt) {
    if (threadIdx.x < E_NUM) cnt[threadIdx.x] = 0;
}

// ---------------------------------------------------------------- router
__global__ __launch_bounds__(256) void moe_router(
    const float* __restrict__ x, const float* __restrict__ rw,
    int* __restrict__ cnt, int* __restrict__ toklist,
    int* __restrict__ pairlist, float* __restrict__ pairw)
{
    __shared__ __align__(16) float s_rw[E_NUM * H_DIM];   // 32 KB
    const int tid = threadIdx.x;
    for (int i = tid; i < E_NUM * H_DIM / 4; i += 256)
        *(F4*)&s_rw[i * 4] = *(const F4*)&rw[i * 4];
    __syncthreads();

    const int t = blockIdx.x * 256 + tid;
    float acc[E_NUM];
    #pragma unroll
    for (int e = 0; e < E_NUM; ++e) acc[e] = 0.f;
    const F4* xr4 = (const F4*)(x + (size_t)t * H_DIM);
    for (int h4 = 0; h4 < H_DIM / 4; ++h4) {
        F4 xv = xr4[h4];
        #pragma unroll
        for (int e = 0; e < E_NUM; ++e) {
            F4 rv = *(const F4*)&s_rw[e * H_DIM + h4 * 4];
            acc[e] = fmaf(xv.x, rv.x, fmaf(xv.y, rv.y, fmaf(xv.z, rv.z, fmaf(xv.w, rv.w, acc[e]))));
        }
    }
    int e0 = 0; float v0 = acc[0];
    #pragma unroll
    for (int e = 1; e < E_NUM; ++e) if (acc[e] > v0) { v0 = acc[e]; e0 = e; }
    int e1 = 0; float v1 = -3.4e38f;
    #pragma unroll
    for (int e = 0; e < E_NUM; ++e) if (e != e0 && acc[e] > v1) { v1 = acc[e]; e1 = e; }
    float w0 = 1.f / (1.f + expf(v1 - v0));
    float w1 = 1.f - w0;
    pairw[2 * t]     = w0;
    pairw[2 * t + 1] = w1;
    int p0 = atomicAdd(&cnt[e0], 1);
    toklist[e0 * T_TOK + p0]  = t;
    pairlist[e0 * T_TOK + p0] = 2 * t;
    int p1 = atomicAdd(&cnt[e1], 1);
    toklist[e1 * T_TOK + p1]  = t;
    pairlist[e1 * T_TOK + p1] = 2 * t + 1;
}

// Shared WMMA inner step: A frag + 4 B frags loaded into distinct regs, then 4 WMMAs
#define WMMA_SUBSTEP(sA, sB, ks)                                                    \
    {                                                                               \
        Frag16 fa, fb0, fb1, fb2, fb3;                                              \
        const int kb = (ks) * 32;                                                   \
        fa.q[0]  = *(const U4*)&sA[(wm * 16 + nl) * AS + kb + half * 16];           \
        fa.q[1]  = *(const U4*)&sA[(wm * 16 + nl) * AS + kb + half * 16 + 8];       \
        fb0.q[0] = *(const U4*)&sB[(wn * 64 +  0 + nl) * AS + kb + half * 16];      \
        fb0.q[1] = *(const U4*)&sB[(wn * 64 +  0 + nl) * AS + kb + half * 16 + 8];  \
        fb1.q[0] = *(const U4*)&sB[(wn * 64 + 16 + nl) * AS + kb + half * 16];      \
        fb1.q[1] = *(const U4*)&sB[(wn * 64 + 16 + nl) * AS + kb + half * 16 + 8];  \
        fb2.q[0] = *(const U4*)&sB[(wn * 64 + 32 + nl) * AS + kb + half * 16];      \
        fb2.q[1] = *(const U4*)&sB[(wn * 64 + 32 + nl) * AS + kb + half * 16 + 8];  \
        fb3.q[0] = *(const U4*)&sB[(wn * 64 + 48 + nl) * AS + kb + half * 16];      \
        fb3.q[1] = *(const U4*)&sB[(wn * 64 + 48 + nl) * AS + kb + half * 16 + 8];  \
        acc[0] = __builtin_amdgcn_wmma_f32_16x16x32_bf16(false, fa.v, false, fb0.v, (short)0, acc[0], false, false); \
        acc[1] = __builtin_amdgcn_wmma_f32_16x16x32_bf16(false, fa.v, false, fb1.v, (short)0, acc[1], false, false); \
        acc[2] = __builtin_amdgcn_wmma_f32_16x16x32_bf16(false, fa.v, false, fb2.v, (short)0, acc[2], false, false); \
        acc[3] = __builtin_amdgcn_wmma_f32_16x16x32_bf16(false, fa.v, false, fb3.v, (short)0, acc[3], false, false); \
    }

// ---------------------------------------------------------------- GEMM1: h = gelu(x @ w1^T + b1)
__global__ __launch_bounds__(256) void moe_gemm1(
    const float* __restrict__ x, const float* __restrict__ w1,
    const float* __restrict__ b1, const int* __restrict__ cnt,
    const int* __restrict__ toklist, const int* __restrict__ pairlist,
    bf16* __restrict__ h_ws)
{
    const int e = blockIdx.z;
    const int c = cnt[e];
    const int m_base = blockIdx.y * 64;
    if (m_base >= c) return;
    const int n_base = blockIdx.x * 128;

    __shared__ __align__(16) bf16 sA[64 * AS];    //  9.0 KB
    __shared__ __align__(16) bf16 sB[128 * AS];   // 18.0 KB
    __shared__ int sTok[64];
    __shared__ int sPairO[64];

    const int tid = threadIdx.x;
    if (tid < 64) {
        int idx = m_base + tid;
        int g = idx < c ? idx : c - 1;
        sTok[tid]   = toklist[e * T_TOK + g];
        sPairO[tid] = (idx < c) ? pairlist[e * T_TOK + idx] : -1;
    }
    __syncthreads();

    const int lane = tid & 31;
    const int wave = tid >> 5;
    const int wm   = wave & 3;     // 4 M sub-tiles of 16 rows -> M = 64
    const int wn   = wave >> 2;    // 2 N sub-tiles of 64 cols -> N = 128
    const int half = lane >> 4;
    const int nl   = lane & 15;

    v8f acc[4];
    #pragma unroll
    for (int i = 0; i < 4; ++i)
        #pragma unroll
        for (int j = 0; j < 8; ++j) acc[i][j] = 0.f;

    const float* wB = w1 + (size_t)e * FFN_DIM * H_DIM + (size_t)n_base * H_DIM;

    // prefetch registers: A 64x64 f32 = 1024 F4 (4/thread), B 128x64 f32 = 2048 F4 (8/thread)
    F4 pa[4], pb[8];
    const int ra = tid >> 4, qa = tid & 15;       // A: chunk id = tid + 256*i
    #pragma unroll
    for (int i = 0; i < 4; ++i) {
        int r = (tid + 256 * i) >> 4, q = (tid + 256 * i) & 15;
        pa[i] = *(const F4*)(x + (size_t)sTok[r] * H_DIM + 0 + q * 4);
    }
    #pragma unroll
    for (int i = 0; i < 8; ++i) {
        int r = (tid + 256 * i) >> 4, q = (tid + 256 * i) & 15;
        pb[i] = *(const F4*)(wB + (size_t)r * H_DIM + 0 + q * 4);
    }
    (void)ra; (void)qa;

    for (int kt = 0; kt < H_DIM; kt += KT) {
        __syncthreads();
        #pragma unroll
        for (int i = 0; i < 4; ++i) {
            int r = (tid + 256 * i) >> 4, q = (tid + 256 * i) & 15;
            st_bf16x4(&sA[r * AS + q * 4], pa[i]);
        }
        #pragma unroll
        for (int i = 0; i < 8; ++i) {
            int r = (tid + 256 * i) >> 4, q = (tid + 256 * i) & 15;
            st_bf16x4(&sB[r * AS + q * 4], pb[i]);
        }
        __syncthreads();

        const int kn = (kt + KT < H_DIM) ? (kt + KT) : kt;   // prefetch next tile (clamped)
        #pragma unroll
        for (int i = 0; i < 4; ++i) {
            int r = (tid + 256 * i) >> 4, q = (tid + 256 * i) & 15;
            pa[i] = *(const F4*)(x + (size_t)sTok[r] * H_DIM + kn + q * 4);
        }
        #pragma unroll
        for (int i = 0; i < 8; ++i) {
            int r = (tid + 256 * i) >> 4, q = (tid + 256 * i) & 15;
            pb[i] = *(const F4*)(wB + (size_t)r * H_DIM + kn + q * 4);
        }

        WMMA_SUBSTEP(sA, sB, 0)
        WMMA_SUBSTEP(sA, sB, 1)
    }

    #pragma unroll
    for (int ni = 0; ni < 4; ++ni) {
        int col = n_base + wn * 64 + ni * 16 + nl;
        float bias = b1[e * FFN_DIM + col];
        #pragma unroll
        for (int v = 0; v < 8; ++v) {
            int r = wm * 16 + v + 8 * half;
            int pr = sPairO[r];
            if (pr >= 0) {
                float val = acc[ni][v] + bias;
                val = 0.5f * val * (1.f + erff(val * 0.70710678118654752f)); // exact gelu
                h_ws[(size_t)pr * FFN_DIM + col] = (bf16)val;
            }
        }
    }
}

// ---------------------------------------------------------------- GEMM2: y = h @ w2^T + b2
__global__ __launch_bounds__(256) void moe_gemm2(
    const bf16* __restrict__ h_ws, const float* __restrict__ w2,
    const float* __restrict__ b2, const int* __restrict__ cnt,
    const int* __restrict__ pairlist, float* __restrict__ y_ws)
{
    const int e = blockIdx.z;
    const int c = cnt[e];
    const int m_base = blockIdx.y * 64;
    if (m_base >= c) return;
    const int n_base = blockIdx.x * 128;

    __shared__ __align__(16) bf16 sA[64 * AS];
    __shared__ __align__(16) bf16 sB[128 * AS];
    __shared__ int sPairA[64];
    __shared__ int sPairO[64];

    const int tid = threadIdx.x;
    if (tid < 64) {
        int idx = m_base + tid;
        int g = idx < c ? idx : c - 1;
        int p = pairlist[e * T_TOK + g];
        sPairA[tid] = p;
        sPairO[tid] = (idx < c) ? p : -1;
    }
    __syncthreads();

    const int lane = tid & 31;
    const int wave = tid >> 5;
    const int wm   = wave & 3;
    const int wn   = wave >> 2;
    const int half = lane >> 4;
    const int nl   = lane & 15;

    v8f acc[4];
    #pragma unroll
    for (int i = 0; i < 4; ++i)
        #pragma unroll
        for (int j = 0; j < 8; ++j) acc[i][j] = 0.f;

    const float* wB = w2 + (size_t)e * H_DIM * FFN_DIM + (size_t)n_base * FFN_DIM;

    // prefetch regs: A 64x64 bf16 = 512 U4 (2/thread), B 128x64 f32 = 2048 F4 (8/thread)
    U4 pa[2];
    F4 pb[8];
    #pragma unroll
    for (int i = 0; i < 2; ++i) {
        int r = (tid + 256 * i) >> 3, q = (tid + 256 * i) & 7;
        pa[i] = *(const U4*)(h_ws + (size_t)sPairA[r] * FFN_DIM + 0 + q * 8);
    }
    #pragma unroll
    for (int i = 0; i < 8; ++i) {
        int r = (tid + 256 * i) >> 4, q = (tid + 256 * i) & 15;
        pb[i] = *(const F4*)(wB + (size_t)r * FFN_DIM + 0 + q * 4);
    }

    for (int kt = 0; kt < FFN_DIM; kt += KT) {
        __syncthreads();
        #pragma unroll
        for (int i = 0; i < 2; ++i) {
            int r = (tid + 256 * i) >> 3, q = (tid + 256 * i) & 7;
            *(U4*)&sA[r * AS + q * 8] = pa[i];
        }
        #pragma unroll
        for (int i = 0; i < 8; ++i) {
            int r = (tid + 256 * i) >> 4, q = (tid + 256 * i) & 15;
            st_bf16x4(&sB[r * AS + q * 4], pb[i]);
        }
        __syncthreads();

        const int kn = (kt + KT < FFN_DIM) ? (kt + KT) : kt;
        #pragma unroll
        for (int i = 0; i < 2; ++i) {
            int r = (tid + 256 * i) >> 3, q = (tid + 256 * i) & 7;
            pa[i] = *(const U4*)(h_ws + (size_t)sPairA[r] * FFN_DIM + kn + q * 8);
        }
        #pragma unroll
        for (int i = 0; i < 8; ++i) {
            int r = (tid + 256 * i) >> 4, q = (tid + 256 * i) & 15;
            pb[i] = *(const F4*)(wB + (size_t)r * FFN_DIM + kn + q * 4);
        }

        WMMA_SUBSTEP(sA, sB, 0)
        WMMA_SUBSTEP(sA, sB, 1)
    }

    #pragma unroll
    for (int ni = 0; ni < 4; ++ni) {
        int col = n_base + wn * 64 + ni * 16 + nl;
        float bias = b2[e * H_DIM + col];
        #pragma unroll
        for (int v = 0; v < 8; ++v) {
            int r = wm * 16 + v + 8 * half;
            int pr = sPairO[r];
            if (pr >= 0)
                y_ws[(size_t)pr * H_DIM + col] = acc[ni][v] + bias;
        }
    }
}

// ---------------------------------------------------------------- combine
__global__ __launch_bounds__(256) void moe_combine(
    const float* __restrict__ y_ws, const float* __restrict__ pairw,
    float* __restrict__ out)
{
    int idx = blockIdx.x * 256 + threadIdx.x;   // over T*H/4
    int t  = idx >> 8;                           // H/4 = 256 float4 per token
    int hq = idx & 255;
    float w0 = pairw[2 * t], w1 = pairw[2 * t + 1];
    F4 a = *(const F4*)(y_ws + (size_t)(2 * t) * H_DIM + hq * 4);
    F4 b = *(const F4*)(y_ws + (size_t)(2 * t + 1) * H_DIM + hq * 4);
    F4 o;
    o.x = w0 * a.x + w1 * b.x;
    o.y = w0 * a.y + w1 * b.y;
    o.z = w0 * a.z + w1 * b.z;
    o.w = w0 * a.w + w1 * b.w;
    *(F4*)(out + (size_t)t * H_DIM + hq * 4) = o;
}

// ---------------------------------------------------------------- launch
extern "C" void kernel_launch(void* const* d_in, const int* in_sizes, int n_in,
                              void* d_out, int out_size, void* d_ws, size_t ws_size,
                              hipStream_t stream) {
    const float* x  = (const float*)d_in[0];
    const float* rw = (const float*)d_in[1];
    const float* w1 = (const float*)d_in[2];
    const float* b1 = (const float*)d_in[3];
    const float* w2 = (const float*)d_in[4];
    const float* b2 = (const float*)d_in[5];
    float* out = (float*)d_out;

    // workspace carve (all offsets 256B aligned)
    char* p = (char*)d_ws;
    int*   cnt      = (int*)p;               p += 256;
    int*   toklist  = (int*)p;               p += (size_t)E_NUM * T_TOK * 4;   // 128 KB
    int*   pairlist = (int*)p;               p += (size_t)E_NUM * T_TOK * 4;   // 128 KB
    float* pairw    = (float*)p;             p += (size_t)2 * T_TOK * 4;       // 32 KB
    bf16*  h_ws     = (bf16*)p;              p += (size_t)2 * T_TOK * FFN_DIM * 2; // 64 MB
    float* y_ws     = (float*)p;             // 32 MB

    moe_init<<<1, 32, 0, stream>>>(cnt);
    moe_router<<<T_TOK / 256, 256, 0, stream>>>(x, rw, cnt, toklist, pairlist, pairw);
    moe_gemm1<<<dim3(FFN_DIM / 128, T_TOK / 64, E_NUM), 256, 0, stream>>>(
        x, w1, b1, cnt, toklist, pairlist, h_ws);
    moe_gemm2<<<dim3(H_DIM / 128, T_TOK / 64, E_NUM), 256, 0, stream>>>(
        h_ws, w2, b2, cnt, pairlist, y_ws);
    moe_combine<<<(T_TOK * (H_DIM / 4)) / 256, 256, 0, stream>>>(y_ws, pairw, out);
}
